// ResMamba_63720134803769
// MI455X (gfx1250) — matrix operand order repaired and checked by reading, not compiled, not run on hardware
//
#include <hip/hip_runtime.h>
#include <hip/hip_bf16.h>

// ---------------------------------------------------------------------------
// Mamba2 block forward on gfx1250 (MI455X, wave32, WMMA).
//   B=32, T=250, D_MODEL=240, D_INNER=480, D_STATE=64, HEADDIM=60, NHEADS=8
//   CONV_DIM=608, D_IN_PROJ=1096
// Pipeline:
//   0) pack W_in / W_out into WMMA fragment-major bf16, n-tiles padded to the
//      block coverage (16 tiles) so the MAC loop is branch-free
//   1) in-proj GEMM  (bf16 WMMA, f32 accum, 16x64 per wave)
//   2) depthwise causal conv(4) + SiLU
//   3) dt softplus / dA = exp(-exp(A_log)*dt)
//   4) selective scan: 1 block per (b,h), thread=p, state[64] in VGPRs
//   5) gate (y*silu(z)) + RMSNorm -> padded bf16 activations
//   6) out-proj GEMM (bf16 WMMA) + fused SELU -> d_out
// ---------------------------------------------------------------------------

#define TT      250
#define DMODEL  240
#define DINNER  480
#define DSTATE  64
#define NHEADS  8
#define HEADDIM 60
#define CONVDIM 608
#define DINPROJ 1096
#define BATCH   32
#define ROWS    (BATCH * TT)         // 8000 rows, multiple of 16

#define KPAD1   256                  // D_MODEL padded to 32
#define KPAD2   512                  // D_INNER padded to 32
#define NT1P    80                   // ceil(69/16)*16 padded n-tiles, in-proj
#define NT2P    16                   // ceil(15/16)*16 padded n-tiles, out-proj
#define KT1     (KPAD1 / 32)         // 8  k-tiles
#define KT2     (KPAD2 / 32)         // 16 k-tiles

typedef __attribute__((ext_vector_type(16))) __bf16 v16bf;
typedef __attribute__((ext_vector_type(8)))  __bf16 v8bf;
typedef __attribute__((ext_vector_type(8)))  float  v8f;

// ---------------------------------------------------------------------------
// Pack weight W[K x N] (f32, row-major) into WMMA B-fragment-major bf16:
//   Wp[kt][nt][lane][e],  n = nt*16 + (lane&15),  k = kt*32 + (lane>>4)*8 + (e&7) + (e>>3)*16
// Zero-padded outside [K,N] (including whole padded n-tiles).
// ---------------------------------------------------------------------------
template<int K, int N, int KT, int NTP>
__global__ void pack_w_kernel(const float* __restrict__ W, __bf16* __restrict__ Wp)
{
    const int idx = blockIdx.x * 256 + threadIdx.x;
    if (idx >= KT * NTP * 512) return;
    const int e    = idx & 15;
    const int lane = (idx >> 4) & 31;
    const int nt   = (idx >> 9) % NTP;
    const int kt   = idx / (512 * NTP);
    const int n = nt * 16 + (lane & 15);
    const int k = kt * 32 + (lane >> 4) * 8 + (e & 7) + (e >> 3) * 16;
    Wp[idx] = (k < K && n < N) ? (__bf16)W[(size_t)k * N + n] : (__bf16)0.0f;
}

// x [ROWS x DMODEL] f32 -> [ROWS x KPAD1] bf16, zero-padded K
__global__ void pack_x_kernel(const float* __restrict__ X, __bf16* __restrict__ Xp)
{
    const int idx = blockIdx.x * 256 + threadIdx.x;   // ROWS*KPAD1
    if (idx >= ROWS * KPAD1) return;
    const int k = idx & (KPAD1 - 1);
    const int m = idx >> 8;
    Xp[idx] = (k < DMODEL) ? (__bf16)X[(size_t)m * DMODEL + k] : (__bf16)0.0f;
}

// ---------------------------------------------------------------------------
// WMMA GEMM:  C[M x N] = A[M x KPAD]bf16 * Bp (fragment-major bf16, NTP tiles).
// One wave -> 16 rows x 4 consecutive 16-col tiles (4 accumulators).
// Block = 4 waves = 16 rows x 16 n-tiles; NTP is a multiple of 16, so the
// fully-unrolled MAC loop has NO branches and NO exec masking.  Real-size
// checks (N, realTiles) exist only in the store epilogue.
// Fragment layouts per CDNA5 ISA 7.12.2 (wave32):
//   A 16x32 bf16 : lane&15 = M; lanes0-15 K{0..7,16..23}, lanes16-31 +8.
//   B 32x16 bf16 : lane&15 = N; lanes0-15 K{0..7,16..23}, lanes16-31 +8.
//   C/D 16x16 f32: 8 VGPRs, N = lane&15, M = r + 8*(lane>>4).
// ---------------------------------------------------------------------------
template<int KPAD, int NTP, int N, bool SELU_EPI>
__global__ __launch_bounds__(128) void wmma_gemm_kernel(
    const __bf16* __restrict__ A,     // [M x KPAD]
    const __bf16* __restrict__ Bp,    // [KPAD/32][NTP][32][16]
    float* __restrict__ C, int M)
{
    constexpr int NTREAL = (N + 15) / 16;
    const int lane   = threadIdx.x & 31;
    const int wave   = threadIdx.x >> 5;
    const int nTile0 = (blockIdx.x * 4 + wave) * 4;   // 4 subtiles per wave
    const int mTile  = blockIdx.y;

    const int hi   = lane >> 4;
    const int mRow = mTile * 16 + (lane & 15);
    const __bf16* arow = A + (size_t)mRow * KPAD;

    v8f acc[4] = {{}, {}, {}, {}};
#pragma unroll
    for (int kk = 0; kk < KPAD; kk += 32) {
        const v8bf alo = *(const v8bf*)(arow + kk + hi * 8);
        const v8bf ahi = *(const v8bf*)(arow + kk + 16 + hi * 8);
        const v16bf a = __builtin_shufflevector(alo, ahi,
            0, 1, 2, 3, 4, 5, 6, 7, 8, 9, 10, 11, 12, 13, 14, 15);
        const __bf16* bbase =
            Bp + ((size_t)(kk >> 5) * NTP + nTile0) * 512 + lane * 16;
#pragma unroll
        for (int s = 0; s < 4; ++s) {
            const v16bf b = *(const v16bf*)(bbase + s * 512);
            acc[s] = __builtin_amdgcn_wmma_f32_16x16x32_bf16(
                         false, a, false, b, (short)0, acc[s],
                         false, false);
        }
    }

#pragma unroll
    for (int s = 0; s < 4; ++s) {
        const int nT = nTile0 + s;
        if (nT < NTREAL) {                       // skip padded tiles (store only)
            const int nCol = nT * 16 + (lane & 15);
            if (nCol < N) {                      // ragged last real tile
#pragma unroll
                for (int r = 0; r < 8; ++r) {
                    const int m = mTile * 16 + r + hi * 8;
                    float v = acc[s][r];
                    if (SELU_EPI) {
                        v = 1.0507009873554805f *
                            (v > 0.0f ? v
                                      : 1.6732632423543772f *
                                            (__expf(v) - 1.0f));
                    }
                    C[(size_t)m * N + nCol] = v;
                }
            }
        }
    }
}

// ---------------------------------------------------------------------------
// Depthwise causal conv width-4 + SiLU.  xBC lives at zxbcdt[..., 480:1088].
// ---------------------------------------------------------------------------
__global__ void conv_silu_kernel(const float* __restrict__ zxbcdt,
                                 const float* __restrict__ conv_w,
                                 const float* __restrict__ conv_b,
                                 float* __restrict__ xconv)
{
    const size_t idx = (size_t)blockIdx.x * blockDim.x + threadIdx.x;
    if (idx >= (size_t)ROWS * CONVDIM) return;
    const int c     = (int)(idx % CONVDIM);
    const size_t bt = idx / CONVDIM;
    const int t     = (int)(bt % TT);
    const size_t b  = bt / TT;

    float acc = conv_b[c];
#pragma unroll
    for (int j = 0; j < 4; ++j) {
        const int tt = t - 3 + j;
        if (tt >= 0)
            acc += zxbcdt[((size_t)b * TT + tt) * DINPROJ + DINNER + c] *
                   conv_w[c * 4 + j];
    }
    xconv[idx] = acc / (1.0f + __expf(-acc));    // SiLU
}

// ---------------------------------------------------------------------------
// dt = softplus(dt_raw + dt_bias); dA = exp(-exp(A_log)*dt)
// ---------------------------------------------------------------------------
__global__ void dtda_kernel(const float* __restrict__ zxbcdt,
                            const float* __restrict__ dt_bias,
                            const float* __restrict__ A_log,
                            float* __restrict__ dt_soft,
                            float* __restrict__ dA)
{
    const int idx = blockIdx.x * blockDim.x + threadIdx.x;   // ROWS*NHEADS
    if (idx >= ROWS * NHEADS) return;
    const int h  = idx & 7;
    const int bt = idx >> 3;
    const float x  = zxbcdt[(size_t)bt * DINPROJ + (DINPROJ - NHEADS) + h]
                   + dt_bias[h];
    const float sp = (x > 20.0f) ? x : log1pf(__expf(x));
    dt_soft[idx] = sp;
    dA[idx]      = __expf(-__expf(A_log[h]) * sp);
}

// ---------------------------------------------------------------------------
// Selective scan.  One 64-thread block per (b,h).  Thread p (<60) keeps
// state[p, 0..63] in 64 VGPRs; B/C staged in LDS per step; y contraction and
// state update are intra-thread FMA chains (no cross-lane reductions).
// ---------------------------------------------------------------------------
__global__ __launch_bounds__(64) void scan_kernel(
    const float* __restrict__ xconv,
    const float* __restrict__ dt_soft,
    const float* __restrict__ dA,
    const float* __restrict__ Dvec,
    float* __restrict__ yscan)
{
    const int bh = blockIdx.x;          // 0..255
    const int b  = bh >> 3;
    const int h  = bh & 7;
    const int p  = threadIdx.x;         // 0..63 (p>=60 only helps load B/C)

    __shared__ float sB[DSTATE];
    __shared__ float sC[DSTATE];

    float state[DSTATE];
#pragma unroll
    for (int n = 0; n < DSTATE; ++n) state[n] = 0.0f;
    const float Dh = Dvec[h];

    for (int t = 0; t < TT; ++t) {
        const size_t bt  = (size_t)b * TT + t;
        const float* row = xconv + bt * CONVDIM;
        sB[p] = row[DINNER + p];
        sC[p] = row[DINNER + DSTATE + p];
        __syncthreads();

        if (p < HEADDIM) {
            const float dtv = dt_soft[bt * NHEADS + h];
            const float dAv = dA[bt * NHEADS + h];
            const float xv  = row[h * HEADDIM + p];
            const float dtx = dtv * xv;
            float y = 0.0f;
#pragma unroll
            for (int n = 0; n < DSTATE; ++n) {
                state[n] = state[n] * dAv + dtx * sB[n];
                y = fmaf(state[n], sC[n], y);
            }
            yscan[bt * DINNER + h * HEADDIM + p] = y + Dh * xv;
        }
        __syncthreads();
    }
}

// ---------------------------------------------------------------------------
// Gate by silu(z), RMSNorm with norm_w; write padded bf16 rows [ROWS x KPAD2]
// for the out-proj GEMM.  One 256-thread block per row (480 real elements).
// ---------------------------------------------------------------------------
__global__ __launch_bounds__(256) void gatenorm_kernel(
    const float* __restrict__ zxbcdt,
    const float* __restrict__ norm_w,
    const float* __restrict__ yscan,
    __bf16* __restrict__ yn)            // [ROWS x KPAD2] bf16, zero-padded
{
    const int row = blockIdx.x;          // 0..7999
    const int tid = threadIdx.x;         // 256
    const float* zrow = zxbcdt + (size_t)row * DINPROJ;   // z = [..., :480]
    const float* yrow = yscan + (size_t)row * DINNER;
    __bf16* orow = yn + (size_t)row * KPAD2;

    const int i0 = tid, i1 = tid + 256;
    float v0 = 0.0f, v1 = 0.0f;
    {
        const float z0 = zrow[i0];
        v0 = yrow[i0] * (z0 / (1.0f + __expf(-z0)));
    }
    if (i1 < DINNER) {
        const float z1 = zrow[i1];
        v1 = yrow[i1] * (z1 / (1.0f + __expf(-z1)));
    }

    __shared__ float red[256];
    red[tid] = v0 * v0 + v1 * v1;
    __syncthreads();
#pragma unroll
    for (int s = 128; s > 0; s >>= 1) {
        if (tid < s) red[tid] += red[tid + s];
        __syncthreads();
    }
    const float scale = rsqrtf(red[0] / (float)DINNER + 1e-5f);

    orow[i0] = (__bf16)(v0 * scale * norm_w[i0]);
    orow[i1] = (i1 < DINNER) ? (__bf16)(v1 * scale * norm_w[i1])
                             : (__bf16)0.0f;
}

// ---------------------------------------------------------------------------
// Launch
// ---------------------------------------------------------------------------
static inline size_t align_up(size_t v) { return (v + 255) & ~(size_t)255; }

extern "C" void kernel_launch(void* const* d_in, const int* in_sizes, int n_in,
                              void* d_out, int out_size, void* d_ws, size_t ws_size,
                              hipStream_t stream)
{
    (void)in_sizes; (void)n_in; (void)out_size; (void)ws_size;
    const float* x       = (const float*)d_in[0];   // [8000,240]
    const float* W_in    = (const float*)d_in[1];   // [240,1096]
    const float* conv_w  = (const float*)d_in[2];   // [608,4]
    const float* conv_b  = (const float*)d_in[3];   // [608]
    const float* dt_bias = (const float*)d_in[4];   // [8]
    const float* A_log   = (const float*)d_in[5];   // [8]
    const float* Dvec    = (const float*)d_in[6];   // [8]
    const float* norm_w  = (const float*)d_in[7];   // [480]
    const float* W_out   = (const float*)d_in[8];   // [480,240]
    float* out = (float*)d_out;                     // [8000,240]

    // Workspace carve (~84 MB)
    char* ws = (char*)d_ws;
    size_t off = 0;
    float* zxbcdt  = (float*)(ws + off); off = align_up(off + (size_t)ROWS * DINPROJ * 4);
    float* xconv   = (float*)(ws + off); off = align_up(off + (size_t)ROWS * CONVDIM * 4);
    float* dt_soft = (float*)(ws + off); off = align_up(off + (size_t)ROWS * NHEADS * 4);
    float* dA      = (float*)(ws + off); off = align_up(off + (size_t)ROWS * NHEADS * 4);
    float* yscan   = (float*)(ws + off); off = align_up(off + (size_t)ROWS * DINNER * 4);
    __bf16* x_bf   = (__bf16*)(ws + off); off = align_up(off + (size_t)ROWS * KPAD1 * 2);
    __bf16* win_p  = (__bf16*)(ws + off); off = align_up(off + (size_t)KT1 * NT1P * 512 * 2);
    __bf16* yn_bf  = (__bf16*)(ws + off); off = align_up(off + (size_t)ROWS * KPAD2 * 2);
    __bf16* wout_p = (__bf16*)(ws + off); off = align_up(off + (size_t)KT2 * NT2P * 512 * 2);

    // 0) pack inputs to bf16 fragment-friendly layouts
    pack_x_kernel<<<(ROWS * KPAD1 + 255) / 256, 256, 0, stream>>>(x, x_bf);
    pack_w_kernel<DMODEL, DINPROJ, KT1, NT1P>
        <<<(KT1 * NT1P * 512 + 255) / 256, 256, 0, stream>>>(W_in, win_p);
    pack_w_kernel<DINNER, DMODEL, KT2, NT2P>
        <<<(KT2 * NT2P * 512 + 255) / 256, 256, 0, stream>>>(W_out, wout_p);

    // 1) in-proj GEMM: [8000,256]bf16 @ packed W_in -> zxbcdt f32
    {
        dim3 grid(NT1P / 16, ROWS / 16);                  // (5, 500)
        wmma_gemm_kernel<KPAD1, NT1P, DINPROJ, false>
            <<<grid, 128, 0, stream>>>(x_bf, win_p, zxbcdt, ROWS);
    }
    // 2) conv + SiLU
    {
        const size_t n = (size_t)ROWS * CONVDIM;
        conv_silu_kernel<<<(unsigned)((n + 255) / 256), 256, 0, stream>>>(
            zxbcdt, conv_w, conv_b, xconv);
    }
    // 3) dt / dA
    dtda_kernel<<<(ROWS * NHEADS + 255) / 256, 256, 0, stream>>>(
        zxbcdt, dt_bias, A_log, dt_soft, dA);
    // 4) selective scan: 256 (b,h) blocks x 64 threads
    scan_kernel<<<BATCH * NHEADS, 64, 0, stream>>>(xconv, dt_soft, dA, Dvec, yscan);
    // 5) gate + RMSNorm -> padded bf16 activations
    gatenorm_kernel<<<ROWS, 256, 0, stream>>>(zxbcdt, norm_w, yscan, yn_bf);
    // 6) out-proj GEMM + SELU: [8000,512]bf16 @ packed W_out -> d_out
    {
        dim3 grid(NT2P / 16, ROWS / 16);                  // (1, 500)
        wmma_gemm_kernel<KPAD2, NT2P, DMODEL, true>
            <<<grid, 128, 0, stream>>>(yn_bf, wout_p, out, ROWS);
    }
}
